// DiffuseMLP_14869176779093
// MI455X (gfx1250) — matrix operand order repaired and checked by reading, not compiled, not run on hardware
//
#include <hip/hip_runtime.h>

// ---------------------------------------------------------------------------
// DiffuseMLP for MI455X (gfx1250, wave32):
//   1) SGConv propagation in transposed (N x B) layout: f32 L2 atomics
//   2) bf16 conversion of activations + weights
//   3) GEMM1: split-K (16-way) v_wmma_f32_16x16x32_bf16 -> 1024 waves,
//      f32 atomic partial accumulation, bias/ReLU epilogue
//   4) GEMM2: fused single-pass WMMA (operands L2-resident)
//   5) tiny fc (H->C=10) as a VALU dot kernel
// ---------------------------------------------------------------------------

typedef __attribute__((ext_vector_type(16))) __bf16 v16bf;
typedef __attribute__((ext_vector_type(8)))  __bf16 v8bf;
typedef __attribute__((ext_vector_type(8)))  float  v8f;

static constexpr int Bv = 128;
static constexpr int Nv = 16384;
static constexpr int Ev = 1048576;
static constexpr int Hv = 512;
static constexpr int Cv = 10;
static constexpr int KCHUNKS = 16;   // split-K factor for GEMM1

// ---------------- degree / normalization ----------------------------------
__global__ void k_init_deg(float* __restrict__ deg) {
    int n = blockIdx.x * blockDim.x + threadIdx.x;
    if (n < Nv) deg[n] = 1.0f;  // self-loop weight 1
}

__global__ void k_deg_accum(const long long* __restrict__ ei,
                            const float* __restrict__ ew,
                            float* __restrict__ deg) {
    int stride = gridDim.x * blockDim.x;
    for (int e = blockIdx.x * blockDim.x + threadIdx.x; e < Ev; e += stride) {
        int dst = (int)ei[(size_t)Ev + e];
        atomicAdd(deg + dst, ew[e]);
    }
}

__global__ void k_dinv(const float* __restrict__ deg, float* __restrict__ dinv) {
    int n = blockIdx.x * blockDim.x + threadIdx.x;
    if (n < Nv) {
        float d = deg[n];
        dinv[n] = (d > 0.0f) ? rsqrtf(fmaxf(d, 1e-12f)) : 0.0f;
    }
}

// ---------------- transpose x (B x N) -> xT (N x B), init hT with self-loop
__global__ void k_transpose_init(const float* __restrict__ x,
                                 const float* __restrict__ dinv,
                                 float* __restrict__ xT,
                                 float* __restrict__ hT) {
    __shared__ float tile[32][33];
    int nb = blockIdx.x;  // N tile
    int bb = blockIdx.y;  // B tile
    int tx = threadIdx.x; // 0..31
    int ty = threadIdx.y; // 0..7
#pragma unroll
    for (int i = 0; i < 4; ++i) {
        int b = bb * 32 + ty + 8 * i;
        int n = nb * 32 + tx;
        tile[ty + 8 * i][tx] = x[(size_t)b * Nv + n];  // coalesced read along N
    }
    __syncthreads();
#pragma unroll
    for (int i = 0; i < 4; ++i) {
        int n = nb * 32 + ty + 8 * i;
        int b = bb * 32 + tx;
        float v  = tile[tx][ty + 8 * i];
        float di = dinv[n];
        xT[(size_t)n * Bv + b] = v;              // coalesced write along B
        hT[(size_t)n * Bv + b] = v * di * di;    // self-loop contribution
    }
}

// ---------------- edge scatter: one wave per edge, lanes = batch -----------
__global__ void k_scatter(const long long* __restrict__ ei,
                          const float* __restrict__ ew,
                          const float* __restrict__ dinv,
                          const float* __restrict__ xT,
                          float* __restrict__ hT) {
    int gid    = blockIdx.x * blockDim.x + threadIdx.x;
    int lane   = gid & 31;
    int wid    = gid >> 5;
    int nwaves = (gridDim.x * blockDim.x) >> 5;
    for (int e = wid; e < Ev; e += nwaves) {
        int src = (int)ei[e];
        int dst = (int)ei[(size_t)Ev + e];
        float norm = dinv[src] * ew[e] * dinv[dst];
        const float4* xs = (const float4*)(xT + (size_t)src * Bv);
        float*        hd = hT + (size_t)dst * Bv;
        float4 v = xs[lane];                    // 512B per edge, contiguous
        int b = lane * 4;
        atomicAdd(hd + b + 0, v.x * norm);
        atomicAdd(hd + b + 1, v.y * norm);
        atomicAdd(hd + b + 2, v.z * norm);
        atomicAdd(hd + b + 3, v.w * norm);
    }
}

// ---------------- f32 -> bf16 elementwise ----------------------------------
__global__ void k_cvt_bf16(const float* __restrict__ src, __bf16* __restrict__ dst,
                           int count) {
    int stride = gridDim.x * blockDim.x;
    for (int i = blockIdx.x * blockDim.x + threadIdx.x; i < count; i += stride)
        dst[i] = (__bf16)src[i];
}

// ---------------- transpose hT (N x B) -> hA bf16 (B x N) ------------------
__global__ void k_transpose_bf16(const float* __restrict__ hT,
                                 __bf16* __restrict__ hA) {
    __shared__ float tile[32][33];
    int nb = blockIdx.x;  // N tile
    int bb = blockIdx.y;  // B tile
    int tx = threadIdx.x;
    int ty = threadIdx.y;
#pragma unroll
    for (int i = 0; i < 4; ++i) {
        int n = nb * 32 + ty + 8 * i;
        int b = bb * 32 + tx;
        tile[ty + 8 * i][tx] = hT[(size_t)n * Bv + b];  // coalesced along B
    }
    __syncthreads();
#pragma unroll
    for (int i = 0; i < 4; ++i) {
        int b = bb * 32 + ty + 8 * i;
        int n = nb * 32 + tx;
        hA[(size_t)b * Nv + n] = (__bf16)tile[tx][ty + 8 * i];  // coalesced along N
    }
}

// ---------------- helper: load A fragment (16-bit 16x32 layout) ------------
// lanes 0-15 hold K 0..15, lanes 16-31 hold K 16..31;
// each lane reads two contiguous 16B chunks.
__device__ __forceinline__ v16bf load_a_frag(const __bf16* aRow, int k0) {
    const v8bf alo = *(const v8bf*)(aRow + k0);
    const v8bf ahi = *(const v8bf*)(aRow + k0 + 8);
    v16bf a;
#pragma unroll
    for (int i = 0; i < 8; ++i) { a[i] = alo[i]; a[i + 8] = ahi[i]; }
    return a;
}

// ---------------- GEMM1: split-K WMMA, f32 atomic partial accumulation -----
// A: M x K row-major bf16 (activations); Bm: N x K row-major bf16 (weights).
// Wave = (kChunk, mtile, ngrp); computes 16 x 64 partial strip over K/KCHUNKS.
__global__ void k_gemm_wmma_splitk(const __bf16* __restrict__ A,
                                   const __bf16* __restrict__ Bm,
                                   float* __restrict__ outF,
                                   int M, int N, int K) {
    int gid  = blockIdx.x * blockDim.x + threadIdx.x;
    int lane = gid & 31;
    int wid  = gid >> 5;
    int nGroups   = N >> 6;                  // 64-wide N groups
    int tilesPerK = (M >> 4) * nGroups;
    int kc    = wid / tilesPerK;
    int t     = wid % tilesPerK;
    int mtile = t / nGroups;
    int ngrp  = t % nGroups;
    if (kc >= KCHUNKS) return;

    int kLen = K / KCHUNKS;
    int kBeg = kc * kLen;
    int kEnd = kBeg + kLen;

    int l  = lane & 15;
    int hi = lane >> 4;
    const __bf16* aRow = A + (size_t)(mtile * 16 + l) * K + hi * 16;
    int n0 = ngrp * 64;

    v8f c0 = {}, c1 = {}, c2 = {}, c3 = {};
    for (int k0 = kBeg; k0 < kEnd; k0 += 32) {
        __builtin_prefetch(aRow + k0 + 512, 0, 0);  // -> global_prefetch_b8
        v16bf a = load_a_frag(aRow, k0);
        // B fragment: lane holds column n0+t*16+(lane&15); 16 consecutive
        // K values starting at k0 + hi*16 -> one contiguous 32B load.
        const v16bf b0 = *(const v16bf*)(Bm + (size_t)(n0 +  0 + l) * K + k0 + hi * 16);
        const v16bf b1 = *(const v16bf*)(Bm + (size_t)(n0 + 16 + l) * K + k0 + hi * 16);
        const v16bf b2 = *(const v16bf*)(Bm + (size_t)(n0 + 32 + l) * K + k0 + hi * 16);
        const v16bf b3 = *(const v16bf*)(Bm + (size_t)(n0 + 48 + l) * K + k0 + hi * 16);
        c0 = __builtin_amdgcn_wmma_f32_16x16x32_bf16(false, a, false, b0, (short)0, c0, false, false);
        c1 = __builtin_amdgcn_wmma_f32_16x16x32_bf16(false, a, false, b1, (short)0, c1, false, false);
        c2 = __builtin_amdgcn_wmma_f32_16x16x32_bf16(false, a, false, b2, (short)0, c2, false, false);
        c3 = __builtin_amdgcn_wmma_f32_16x16x32_bf16(false, a, false, b3, (short)0, c3, false, false);
    }

    // Partial accumulation: C/D layout -> VGPR r = (row m0+hi*8+r, col n0+t*16+l)
    v8f acc[4] = {c0, c1, c2, c3};
#pragma unroll
    for (int t4 = 0; t4 < 4; ++t4) {
        int col = n0 + t4 * 16 + l;
#pragma unroll
        for (int r = 0; r < 8; ++r) {
            int row = mtile * 16 + hi * 8 + r;
            atomicAdd(outF + (size_t)row * N + col, acc[t4][r]);
        }
    }
}

// seed GEMM1 accumulator with the bias (also un-poisons the buffer)
__global__ void k_fill_bias(const float* __restrict__ bias, float* __restrict__ outF,
                            int count, int N) {
    int i = blockIdx.x * blockDim.x + threadIdx.x;
    if (i < count) outF[i] = bias[i & (N - 1)];
}

// ReLU + bf16 convert epilogue for GEMM1
__global__ void k_relu_cvt_bf16(const float* __restrict__ src, __bf16* __restrict__ dst,
                                int count) {
    int i = blockIdx.x * blockDim.x + threadIdx.x;
    if (i < count) dst[i] = (__bf16)fmaxf(src[i], 0.0f);
}

// ---------------- GEMM2: single-pass WMMA with fused bias+ReLU -------------
__global__ void k_gemm_wmma_fused(const __bf16* __restrict__ A,
                                  const __bf16* __restrict__ Bm,
                                  const float* __restrict__ bias,
                                  float* __restrict__ outF,
                                  int M, int N, int K) {
    int gid  = blockIdx.x * blockDim.x + threadIdx.x;
    int lane = gid & 31;
    int wid  = gid >> 5;
    int nGroups = N >> 6;
    int mtile = wid / nGroups;
    int ngrp  = wid % nGroups;
    if (mtile * 16 >= M) return;

    int l  = lane & 15;
    int hi = lane >> 4;
    const __bf16* aRow = A + (size_t)(mtile * 16 + l) * K + hi * 16;
    int n0 = ngrp * 64;

    v8f c0 = {}, c1 = {}, c2 = {}, c3 = {};
    for (int k0 = 0; k0 < K; k0 += 32) {
        v16bf a = load_a_frag(aRow, k0);
        const v16bf b0 = *(const v16bf*)(Bm + (size_t)(n0 +  0 + l) * K + k0 + hi * 16);
        const v16bf b1 = *(const v16bf*)(Bm + (size_t)(n0 + 16 + l) * K + k0 + hi * 16);
        const v16bf b2 = *(const v16bf*)(Bm + (size_t)(n0 + 32 + l) * K + k0 + hi * 16);
        const v16bf b3 = *(const v16bf*)(Bm + (size_t)(n0 + 48 + l) * K + k0 + hi * 16);
        c0 = __builtin_amdgcn_wmma_f32_16x16x32_bf16(false, a, false, b0, (short)0, c0, false, false);
        c1 = __builtin_amdgcn_wmma_f32_16x16x32_bf16(false, a, false, b1, (short)0, c1, false, false);
        c2 = __builtin_amdgcn_wmma_f32_16x16x32_bf16(false, a, false, b2, (short)0, c2, false, false);
        c3 = __builtin_amdgcn_wmma_f32_16x16x32_bf16(false, a, false, b3, (short)0, c3, false, false);
    }

    v8f acc[4] = {c0, c1, c2, c3};
#pragma unroll
    for (int t = 0; t < 4; ++t) {
        int col = n0 + t * 16 + l;
        float bvv = bias[col];
#pragma unroll
        for (int r = 0; r < 8; ++r) {
            int row = mtile * 16 + hi * 8 + r;
            float v = acc[t][r] + bvv;
            outF[(size_t)row * N + col] = v > 0.0f ? v : 0.0f;
        }
    }
}

// ---------------- final fc (H -> C=10), f32 --------------------------------
__global__ void k_fc(const float* __restrict__ h, const float* __restrict__ Wfc,
                     const float* __restrict__ bfc, float* __restrict__ out) {
    int t = blockIdx.x * blockDim.x + threadIdx.x;
    if (t >= Bv * Cv) return;
    int b = t / Cv, c = t % Cv;
    const float* hb = h   + (size_t)b * Hv;
    const float* wr = Wfc + (size_t)c * Hv;
    float acc = bfc[c];
#pragma unroll 8
    for (int i = 0; i < Hv; ++i) acc = fmaf(hb[i], wr[i], acc);
    out[t] = acc;
}

// ---------------------------------------------------------------------------
static inline size_t align256(size_t x) { return (x + 255) & ~(size_t)255; }

extern "C" void kernel_launch(void* const* d_in, const int* in_sizes, int n_in,
                              void* d_out, int out_size, void* d_ws, size_t ws_size,
                              hipStream_t stream) {
    const float*     x   = (const float*)d_in[0];
    const long long* ei  = (const long long*)d_in[1];   // int64 edge_index (2, E)
    const float*     ew  = (const float*)d_in[2];
    const float*     W1  = (const float*)d_in[3];
    const float*     b1  = (const float*)d_in[4];
    const float*     W2  = (const float*)d_in[5];
    const float*     b2  = (const float*)d_in[6];
    const float*     Wfc = (const float*)d_in[7];
    const float*     bfc = (const float*)d_in[8];
    float*           out = (float*)d_out;

    // workspace carve-up
    char* ws = (char*)d_ws;
    float*  deg   = (float*)ws;  ws += align256((size_t)Nv * 4);
    float*  dinv  = (float*)ws;  ws += align256((size_t)Nv * 4);
    float*  xT    = (float*)ws;  ws += align256((size_t)Nv * Bv * 4);
    float*  hT    = (float*)ws;  ws += align256((size_t)Nv * Bv * 4);
    __bf16* hA    = (__bf16*)ws; ws += align256((size_t)Bv * Nv * 2);
    __bf16* W1b   = (__bf16*)ws; ws += align256((size_t)Hv * Nv * 2);
    __bf16* W2b   = (__bf16*)ws; ws += align256((size_t)Hv * Hv * 2);
    float*  h1f   = (float*)ws;  ws += align256((size_t)Bv * Hv * 4);
    __bf16* h1b   = (__bf16*)ws; ws += align256((size_t)Bv * Hv * 2);
    float*  h2    = (float*)ws;  ws += align256((size_t)Bv * Hv * 4);

    // 1) degrees + normalization
    k_init_deg<<<Nv / 256, 256, 0, stream>>>(deg);
    k_deg_accum<<<1024, 256, 0, stream>>>(ei, ew, deg);
    k_dinv<<<Nv / 256, 256, 0, stream>>>(deg, dinv);

    // 2) transpose x -> xT, seed hT with self-loop term
    k_transpose_init<<<dim3(Nv / 32, Bv / 32), dim3(32, 8), 0, stream>>>(x, dinv, xT, hT);

    // 3) edge scatter (wave per edge, lanes over batch)
    k_scatter<<<2048, 256, 0, stream>>>(ei, ew, dinv, xT, hT);

    // 4) bf16 conversions (weights) and transposed activations
    k_cvt_bf16<<<8192, 256, 0, stream>>>(W1, W1b, Hv * Nv);
    k_cvt_bf16<<<1024, 256, 0, stream>>>(W2, W2b, Hv * Hv);
    k_transpose_bf16<<<dim3(Nv / 32, Bv / 32), dim3(32, 8), 0, stream>>>(hT, hA);

    // 5) GEMM1: split-K x16 -> 1024 waves, seed with bias, relu+cvt epilogue
    k_fill_bias<<<(Bv * Hv) / 256, 256, 0, stream>>>(b1, h1f, Bv * Hv, Hv);
    {
        int waves = (Bv / 16) * (Hv / 64) * KCHUNKS;  // 8 * 8 * 16 = 1024 waves
        k_gemm_wmma_splitk<<<(waves * 32) / 256, 256, 0, stream>>>(
            hA, W1b, h1f, Bv, Hv, Nv);
    }
    k_relu_cvt_bf16<<<(Bv * Hv) / 256, 256, 0, stream>>>(h1f, h1b, Bv * Hv);

    // 6) GEMM2: h2 = relu(h1b @ W2^T + b2) -> f32   (M=128, N=512, K=512)
    {
        int waves = (Bv / 16) * (Hv / 64);  // 64 waves, operands L2-resident
        k_gemm_wmma_fused<<<(waves * 32) / 256, 256, 0, stream>>>(
            h1b, W2b, b2, h2, Bv, Hv, Hv);
    }

    // 7) fc: out = h2 @ Wfc^T + bfc  (128 x 10)
    k_fc<<<(Bv * Cv + 255) / 256, 256, 0, stream>>>(h2, Wfc, bfc, out);
}
